// CrossModalRefinementCell_53171695124721
// MI455X (gfx1250) — matrix operand order repaired and checked by reading, not compiled, not run on hardware
//
#include <hip/hip_runtime.h>

// ---------------------------------------------------------------------------
// CrossModalRefinementCell for MI455X (gfx1250), wave32 + WMMA bf16.
//
// Dominant work: out[i,j,:] = img[j] + relu(pre_txt[i]+pre_img[j]+rb1) @ rw2
//   -> GEMM with M=65536 (i,j) pairs, K=N=1024 => ~137 GFLOP, 256MB fp32 out.
// HBM floor: 256MB / 23.3 TB/s ~= 11us; bf16 WMMA (fp32 accum) keeps the
// matrix path at matrix-engine rates. rw2 (2MB bf16, pre-transposed [N][K])
// and the bf16 pre-activations (0.5MB each) are L2-resident (192MB L2).
// Refine kernel: 64x64 output tile (4 waves x 4 accumulators), software
// pipeline with double-buffered LDS; B chunks stream via
// GLOBAL_LOAD_ASYNC_TO_LDS_B128 (ASYNCcnt) issued a chunk ahead, hid chunks
// are rebuilt in registers with packed bf16 VALU (v_pk_add/max_bf16) and
// ds_store'd after the 8 WMMAs. LDS rows padded to 72 bf16 (36 dwords) so
// 16-row b128 fragment loads are bank-conflict-free. hid is never
// materialized (would be 128MB as bf16).
// ---------------------------------------------------------------------------

#define DIM 1024
#define BS  256
#define KC  64            // k-chunk per pipeline stage (2 WMMA k-steps)
#define NCH (DIM / KC)    // 16 chunks
#define HROW 72           // padded LDS row (bf16): 36 dwords -> conflict-free

typedef __attribute__((ext_vector_type(16))) __bf16 bf16x16;
typedef __attribute__((ext_vector_type(2)))  __bf16 bf16x2;
typedef __attribute__((ext_vector_type(8)))  float  f32x8;

union FragU {             // 32 bytes: two 16B LDS loads -> one WMMA operand
    uint4   q[2];
    bf16x16 v;
};

__device__ __forceinline__ unsigned short f32_to_bf16(float f) {
    unsigned int u = __float_as_uint(f);
    u += 0x7fffu + ((u >> 16) & 1u);     // round-to-nearest-even
    return (unsigned short)(u >> 16);
}

// packed bf16: relu(a + b) on both halves of a dword -> v_pk_add/max_bf16
__device__ __forceinline__ unsigned int relu_add_pk(unsigned int a, unsigned int b) {
    union { unsigned int u; bf16x2 v; } x, y, r;
    x.u = a; y.u = b;
    bf16x2 s = x.v + y.v;
    r.v = __builtin_elementwise_max(s, bf16x2{(__bf16)0.0f, (__bf16)0.0f});
    return r.u;
}

// ---------------------------------------------------------------------------
// fp32 -> bf16 (grid-stride, row-major preserved)
// ---------------------------------------------------------------------------
__global__ void f2bf_kernel(const float* __restrict__ src,
                            unsigned short* __restrict__ dst, int n) {
    for (int i = blockIdx.x * blockDim.x + threadIdx.x; i < n;
         i += gridDim.x * blockDim.x)
        dst[i] = f32_to_bf16(src[i]);
}

// ---------------------------------------------------------------------------
// fp32 [K][N] -> bf16 [N][K] tiled transpose (both sides coalesced)
// ---------------------------------------------------------------------------
__global__ void f2bf_transpose_kernel(const float* __restrict__ src,
                                      unsigned short* __restrict__ dst,
                                      int K, int N) {
    __shared__ float tile[32][33];
    const int k0 = blockIdx.y * 32, n0 = blockIdx.x * 32;
    const int tx = threadIdx.x & 31, ty = threadIdx.x >> 5;   // 8 rows/pass
    for (int r = ty; r < 32; r += 8)
        tile[r][tx] = src[(size_t)(k0 + r) * N + (n0 + tx)];
    __syncthreads();
    for (int r = ty; r < 32; r += 8)
        dst[(size_t)(n0 + r) * K + (k0 + tx)] = f32_to_bf16(tile[tx][r]);
}

// ---------------------------------------------------------------------------
// bf16 WMMA GEMM: act(A[M,K] @ B[K,N] + bias), B pre-transposed as Bt[N][K].
// K = N = 1024. Block: 128 threads (4 waves), 16 x 64 tile; wave w owns
// n-strip w*16. Outputs fp32 C and/or bf16 Cbf (nullable). Minor kernel
// (~0.5 GFLOP per call) -- simple single-buffer loop, padded LDS rows.
// ---------------------------------------------------------------------------
__global__ void gemm_bf16_kernel(const unsigned short* __restrict__ A,
                                 const unsigned short* __restrict__ Bt,
                                 const float* __restrict__ bias,
                                 float* __restrict__ C,
                                 unsigned short* __restrict__ Cbf,
                                 int relu) {
    __shared__ __align__(16) unsigned short Alds[16][1032];   // 516 dw rows
    __shared__ __align__(16) unsigned short Blds[64][40];     // 20 dw rows

    const int tid  = threadIdx.x;
    const int lane = tid & 31;
    const int w    = tid >> 5;
    const int m0   = blockIdx.x * 16;
    const int n0   = blockIdx.y * 64;

    // Stage 16 rows of A (b128-vectorized, coalesced over k).
    for (int idx = tid * 8; idx < 16 * DIM; idx += 128 * 8) {
        int m = idx >> 10, k = idx & (DIM - 1);
        *(uint4*)&Alds[m][k] = *(const uint4*)&A[(size_t)(m0 + m) * DIM + k];
    }
    __syncthreads();

    f32x8 acc = {};
    const int am   = lane & 15;
    const int aoff = (lane >= 16) ? 8 : 0;   // ISA 16-bit A layout (16x32)
    const int bl   = lane & 15;
    const int boff = (lane >= 16) ? 8 : 0;   // ISA 16-bit B layout (32x16)
    const int srow = tid >> 1;               // B-staging row 0..63
    const int skh  = (tid & 1) * 16;         // k-half

    for (int k0 = 0; k0 < DIM; k0 += 32) {
        const uint4* bs =
            (const uint4*)&Bt[(size_t)(n0 + srow) * DIM + k0 + skh];
        *(uint4*)&Blds[srow][skh]     = bs[0];
        *(uint4*)&Blds[srow][skh + 8] = bs[1];
        __syncthreads();

        FragU a, b;
        a.q[0] = *(const uint4*)&Alds[am][k0 + aoff];
        a.q[1] = *(const uint4*)&Alds[am][k0 + aoff + 16];
        b.q[0] = *(const uint4*)&Blds[w * 16 + bl][boff];
        b.q[1] = *(const uint4*)&Blds[w * 16 + bl][boff + 16];

        acc = __builtin_amdgcn_wmma_f32_16x16x32_bf16(
            false, a.v, false, b.v, (short)0, acc, false, false);
        __syncthreads();
    }

    const int   ng = n0 + w * 16 + bl;
    const float bv = bias ? bias[ng] : 0.0f;
#pragma unroll
    for (int v = 0; v < 8; ++v) {
        int   m   = (lane < 16) ? v : (v + 8);
        float val = acc[v] + bv;
        if (relu) val = fmaxf(val, 0.0f);
        size_t o = (size_t)(m0 + m) * DIM + ng;
        if (C)   C[o]   = val;
        if (Cbf) Cbf[o] = f32_to_bf16(val);
    }
}

// ---------------------------------------------------------------------------
// Row softmax over features + multiply by text_features -> bf16
// ---------------------------------------------------------------------------
__global__ void softmax_mul_kernel(const float* __restrict__ Z,
                                   const float* __restrict__ text,
                                   unsigned short* __restrict__ Xbf) {
    __shared__ float red[256];
    const int row = blockIdx.x, tid = threadIdx.x;
    const float* z = Z + (size_t)row * DIM;

    float m = -3.4e38f;
    for (int k = tid; k < DIM; k += 256) m = fmaxf(m, z[k]);
    red[tid] = m; __syncthreads();
    for (int s = 128; s > 0; s >>= 1) {
        if (tid < s) red[tid] = fmaxf(red[tid], red[tid + s]);
        __syncthreads();
    }
    m = red[0]; __syncthreads();

    float sum = 0.f;
    for (int k = tid; k < DIM; k += 256) sum += __expf(z[k] - m);
    red[tid] = sum; __syncthreads();
    for (int s = 128; s > 0; s >>= 1) {
        if (tid < s) red[tid] += red[tid + s];
        __syncthreads();
    }
    const float inv = 1.0f / red[0];

    for (int k = tid; k < DIM; k += 256) {
        float attn = __expf(z[k] - m) * inv;
        Xbf[(size_t)row * DIM + k] =
            f32_to_bf16(text[(size_t)row * DIM + k] * attn);
    }
}

// ---------------------------------------------------------------------------
// Fused pairwise refinement. Block = (i, 64 j's, 64 n's), 128 threads.
//   out[i,j,n] = img[j,n] + sum_k relu(preTxt[i,k]+preImg[j,k]) * rw2[k,n] + rb2[n]
// (rb1 folded into preImg.) Wave w owns m-strip w*16; 4 n-accumulators.
// Pipeline per chunk c: async B loads (chunk c+1) + hid register loads issue
// before the 8 WMMAs; packed-bf16 relu + ds_store after; one barrier.
// ---------------------------------------------------------------------------
__global__ void refine_kernel(const unsigned short* __restrict__ PtxtBf,
                              const unsigned short* __restrict__ PimgBf,
                              const unsigned short* __restrict__ RW2t,  // [n][k]
                              const float* __restrict__ img,
                              const float* __restrict__ rb2,
                              float* __restrict__ out) {
    __shared__ __align__(16) unsigned short Hlds[2][64][HROW];
    __shared__ __align__(16) unsigned short Blds[2][64][HROW];

    const int tid  = threadIdx.x;
    const int lane = tid & 31;
    const int w    = tid >> 5;
    const int n0   = blockIdx.x * 64;
    const int j0   = blockIdx.y * 64;
    const int i    = blockIdx.z;

    const int srow = tid >> 1;           // staging row 0..63
    const int skh  = (tid & 1) * 32;     // k-half of the 64-wide chunk

    f32x8 acc[4] = {};
    const int am   = w * 16 + (lane & 15);
    const int aoff = (lane >= 16) ? 8 : 0;
    const int bl   = lane & 15;
    const int boff = (lane >= 16) ? 8 : 0;

    const unsigned short* ptRow = PtxtBf + (size_t)i * DIM;
    const unsigned short* piRow = PimgBf + (size_t)(j0 + srow) * DIM;
    const unsigned short* bRow  = RW2t + (size_t)(n0 + srow) * DIM;

    uint4 rP[4], rT[4];                  // register staging for hid chunk

    // Async DMA of one 64x64 B chunk directly into LDS (ASYNCcnt-tracked).
    // instruction offset applies to both the global and the LDS address.
    auto issue_asyncB = [&](int buf, int k0) {
        unsigned lb = (unsigned)(size_t)&Blds[buf][srow][skh];
        const unsigned short* gp = &bRow[k0 + skh];
        asm volatile("global_load_async_to_lds_b128 %0, %1, off"
                     :: "v"(lb), "v"(gp) : "memory");
        asm volatile("global_load_async_to_lds_b128 %0, %1, off offset:16"
                     :: "v"(lb), "v"(gp) : "memory");
        asm volatile("global_load_async_to_lds_b128 %0, %1, off offset:32"
                     :: "v"(lb), "v"(gp) : "memory");
        asm volatile("global_load_async_to_lds_b128 %0, %1, off offset:48"
                     :: "v"(lb), "v"(gp) : "memory");
    };
    auto load_stage = [&](int k0) {      // hid inputs -> registers
        const uint4* p = (const uint4*)&piRow[k0 + skh];
        const uint4* t = (const uint4*)&ptRow[k0 + skh];
#pragma unroll
        for (int q = 0; q < 4; ++q) { rP[q] = p[q]; rT[q] = t[q]; }
    };
    auto store_stage = [&](int buf) {    // packed bf16 relu-add -> LDS
#pragma unroll
        for (int q = 0; q < 4; ++q) {
            uint4 h;
            h.x = relu_add_pk(rP[q].x, rT[q].x);
            h.y = relu_add_pk(rP[q].y, rT[q].y);
            h.z = relu_add_pk(rP[q].z, rT[q].z);
            h.w = relu_add_pk(rP[q].w, rT[q].w);
            *(uint4*)&Hlds[buf][srow][skh + q * 8] = h;
        }
    };

    issue_asyncB(0, 0);
    load_stage(0);
    store_stage(0);
    asm volatile("s_wait_asynccnt 0" ::: "memory");
    __syncthreads();

    for (int c = 0; c < NCH; ++c) {
        const int cur = c & 1;
        if (c + 1 < NCH) {                           // prefetch chunk c+1
            issue_asyncB(cur ^ 1, (c + 1) * KC);
            load_stage((c + 1) * KC);
        }

#pragma unroll
        for (int kk = 0; kk < KC; kk += 32) {        // 2 WMMA k-steps
            FragU a;
            a.q[0] = *(const uint4*)&Hlds[cur][am][kk + aoff];
            a.q[1] = *(const uint4*)&Hlds[cur][am][kk + aoff + 16];
#pragma unroll
            for (int s = 0; s < 4; ++s) {
                FragU b;
                b.q[0] = *(const uint4*)&Blds[cur][s * 16 + bl][kk + boff];
                b.q[1] = *(const uint4*)&Blds[cur][s * 16 + bl][kk + boff + 16];
                acc[s] = __builtin_amdgcn_wmma_f32_16x16x32_bf16(
                    false, a.v, false, b.v, (short)0, acc[s], false, false);
            }
        }

        if (c + 1 < NCH) store_stage(cur ^ 1);       // pack + LDS after WMMAs
        asm volatile("s_wait_asynccnt 0" ::: "memory");
        __syncthreads();
    }

    // ---- epilogue: out = img[j] + acc + rb2 (fp32 HBM stream) ----
#pragma unroll
    for (int s = 0; s < 4; ++s) {
        const int   ng = n0 + s * 16 + bl;
        const float b2 = rb2[ng];
#pragma unroll
        for (int v = 0; v < 8; ++v) {
            int m = w * 16 + ((lane < 16) ? v : (v + 8));
            int j = j0 + m;
            out[((size_t)i * BS + j) * DIM + ng] =
                img[(size_t)j * DIM + ng] + acc[s][v] + b2;
        }
    }
}

// ---------------------------------------------------------------------------
extern "C" void kernel_launch(void* const* d_in, const int* in_sizes, int n_in,
                              void* d_out, int out_size, void* d_ws, size_t ws_size,
                              hipStream_t stream) {
    (void)in_sizes; (void)n_in; (void)out_size; (void)ws_size;

    const float* img  = (const float*)d_in[0];
    const float* text = (const float*)d_in[1];
    const float* aw1  = (const float*)d_in[2];
    const float* ab1  = (const float*)d_in[3];
    const float* aw2  = (const float*)d_in[4];
    const float* ab2  = (const float*)d_in[5];
    const float* rw1  = (const float*)d_in[6];
    const float* rb1  = (const float*)d_in[7];
    const float* rw2  = (const float*)d_in[8];
    const float* rb2  = (const float*)d_in[9];
    // d_in[10] = cell_id; setup fixes it to 1 and out_size is BS*BS*DIM,
    // so only the pairwise branch is implemented.
    float* out = (float*)d_out;

    // ---- workspace carve-out (~14 MB) ----
    char* p = (char*)d_ws;
    auto carve = [&](size_t bytes) {
        char* r = p; p += (bytes + 255) & ~size_t(255); return r;
    };
    unsigned short* Tbf    = (unsigned short*)carve((size_t)BS * DIM * 2);
    unsigned short* Ibf    = (unsigned short*)carve((size_t)BS * DIM * 2);
    unsigned short* AW1t   = (unsigned short*)carve((size_t)DIM * DIM * 2);
    unsigned short* AW2t   = (unsigned short*)carve((size_t)DIM * DIM * 2);
    unsigned short* RW1at  = (unsigned short*)carve((size_t)DIM * DIM * 2);
    unsigned short* RW1bt  = (unsigned short*)carve((size_t)DIM * DIM * 2);
    unsigned short* RW2t   = (unsigned short*)carve((size_t)DIM * DIM * 2);
    unsigned short* Hbf    = (unsigned short*)carve((size_t)BS * DIM * 2);
    unsigned short* Xbf    = (unsigned short*)carve((size_t)BS * DIM * 2);
    unsigned short* PimgBf = (unsigned short*)carve((size_t)BS * DIM * 2);
    unsigned short* PtxtBf = (unsigned short*)carve((size_t)BS * DIM * 2);
    float*          Zf     = (float*)carve((size_t)BS * DIM * 4);

    // ---- conversions ----
    f2bf_kernel<<<(BS * DIM) / 256, 256, 0, stream>>>(text, Tbf, BS * DIM);
    f2bf_kernel<<<(BS * DIM) / 256, 256, 0, stream>>>(img,  Ibf, BS * DIM);
    const dim3 gT(DIM / 32, DIM / 32);
    f2bf_transpose_kernel<<<gT, 256, 0, stream>>>(aw1, AW1t, DIM, DIM);
    f2bf_transpose_kernel<<<gT, 256, 0, stream>>>(aw2, AW2t, DIM, DIM);
    f2bf_transpose_kernel<<<gT, 256, 0, stream>>>(rw1, RW1at, DIM, DIM);
    f2bf_transpose_kernel<<<gT, 256, 0, stream>>>(rw1 + (size_t)DIM * DIM,
                                                  RW1bt, DIM, DIM);
    f2bf_transpose_kernel<<<gT, 256, 0, stream>>>(rw2, RW2t, DIM, DIM);

    const dim3 gGemm(BS / 16, DIM / 64);
    // h = relu(text @ aw1 + ab1) -> bf16
    gemm_bf16_kernel<<<gGemm, 128, 0, stream>>>(Tbf, AW1t, ab1, nullptr, Hbf, 1);
    // z = h @ aw2 + ab2 -> fp32 (softmax input)
    gemm_bf16_kernel<<<gGemm, 128, 0, stream>>>(Hbf, AW2t, ab2, Zf, nullptr, 0);
    // text' = text * softmax(z, axis=1) -> bf16
    softmax_mul_kernel<<<BS, 256, 0, stream>>>(Zf, text, Xbf);
    // pre_img = img @ rw1[:D] + rb1 -> bf16 (rb1 folded here)
    gemm_bf16_kernel<<<gGemm, 128, 0, stream>>>(Ibf, RW1at, rb1, nullptr, PimgBf, 0);
    // pre_txt = text' @ rw1[D:] -> bf16
    gemm_bf16_kernel<<<gGemm, 128, 0, stream>>>(Xbf, RW1bt, nullptr, nullptr, PtxtBf, 0);
    // fused pairwise refinement: 137 GFLOP of bf16 WMMA, 256MB fp32 out
    refine_kernel<<<dim3(DIM / 64, BS / 64, BS), 128, 0, stream>>>(
        PtxtBf, PimgBf, RW2t, img, rb2, out);
}
